// DeepAndWide_65060164600286
// MI455X (gfx1250) — compile-verified, matrix operand order checked by reading, not run on hardware
//
#include <hip/hip_runtime.h>
#include <hip/hip_bf16.h>

// ---------------- problem constants (match reference) ----------------
#define BSZ   16384      // batch
#define NFLD  50         // fields
#define EK    16         // embedding dim
#define DIN   (NFLD*EK)  // 800
#define N1    256
#define N2    128

typedef __attribute__((ext_vector_type(16))) _Float16 v16h;
typedef __attribute__((ext_vector_type(8)))  _Float16 v8h;
typedef __attribute__((ext_vector_type(8)))  float    v8f;

// ---------------- 1: zero float region ----------------
__global__ void daw_zero(float* __restrict__ p, long n) {
    long t = (long)blockIdx.x * blockDim.x + threadIdx.x;
    long stride = (long)gridDim.x * blockDim.x;
    for (; t < n; t += stride) p[t] = 0.0f;
}

// ---------------- 2: scatter (wide sum + field-aware sum + counts) ----------------
// one thread per (nonzero, k) pair; k==0 thread also does count + wide term
__global__ void daw_scatter(const int* __restrict__ index,
                            const int* __restrict__ feats,
                            const float* __restrict__ values,
                            const int* __restrict__ fields,
                            const float* __restrict__ weights,
                            const float* __restrict__ emb,
                            float* __restrict__ first,
                            float* __restrict__ cnt,
                            float* __restrict__ h,
                            int nnz) {
    long t = (long)blockIdx.x * blockDim.x + threadIdx.x;
    long total = (long)nnz * EK;
    if (t >= total) return;
    int nz = (int)(t >> 4);
    int k  = (int)(t & 15);
    int b  = index[nz];
    int f  = fields[nz];
    int ft = feats[nz];
    int seg = b * NFLD + f;
    atomicAdd(&h[(size_t)seg * EK + k], emb[(size_t)ft * EK + k]);
    if (k == 0) {
        atomicAdd(&cnt[seg], 1.0f);
        atomicAdd(&first[b], weights[ft] * values[nz]);
    }
}

// ---------------- 3a: f32 -> f16 plain conversion (for W3) ----------------
__global__ void daw_cvt_f16(const float* __restrict__ src,
                            _Float16* __restrict__ dst, int n) {
    int t = blockIdx.x * blockDim.x + threadIdx.x;
    if (t < n) dst[t] = (_Float16)src[t];
}

// ---------------- 3b: pack W [Kd, Nd] f32 row-major into B-fragment layout ----------------
// Packed layout: Wp[((tile * nK + chunk) * 32 + lane) * 16 + e]
// where for lane L (hl=L>>4, ln16=L&15) element e holds
//   W[(chunk*32 + e + 8*((e>>3)+hl)) * Nd + tile*16 + ln16]
// so the GEMM inner loop can fetch each lane's 16 halves as two b128 loads.
__global__ void daw_pack_b(const float* __restrict__ W,
                           _Float16* __restrict__ Wp,
                           int Kd, int Nd) {
    int p = blockIdx.x * blockDim.x + threadIdx.x;
    if (p >= Kd * Nd) return;
    int nK   = Kd >> 5;
    int e    = p & 15;
    int L    = (p >> 4) & 31;
    int rest = p >> 9;
    int c    = rest % nK;
    int tile = rest / nK;
    int hl   = L >> 4;
    int ln16 = L & 15;
    int k = c * 32 + e + 8 * ((e >> 3) + hl);
    int n = tile * 16 + ln16;
    Wp[p] = (_Float16)W[(size_t)k * Nd + n];
}

// ---------------- 4: scatter-mean normalize -> f16 X [B, 800] ----------------
__global__ void daw_normalize(const float* __restrict__ h,
                              const float* __restrict__ cnt,
                              _Float16* __restrict__ X, long total) {
    long t = (long)blockIdx.x * blockDim.x + threadIdx.x;
    if (t >= total) return;
    int b = (int)(t / DIN);
    int f = (int)((t % DIN) >> 4);
    float c = cnt[(size_t)b * NFLD + f];
    c = c > 1.0f ? c : 1.0f;
    X[t] = (_Float16)(h[t] / c);
}

// ---------------- 5: WMMA GEMM, Y = relu(X @ W + b) ----------------
// X: [M, Kd] f16 row-major; Wp: pre-packed B fragments; Y: [M, Nd] f16.
// 4 waves per block; each wave computes a 16x64 output strip:
// one A-fragment is reused across 4 v_wmma_f32_16x16x32_f16 per K-step.
__global__ __launch_bounds__(128)
void daw_gemm_relu_wmma(const _Float16* __restrict__ X,
                        const _Float16* __restrict__ Wp,
                        const float* __restrict__ bptr,
                        _Float16* __restrict__ Y,
                        int M, int Kd, int Nd) {
    const int lane = threadIdx.x & 31;
    const int wave = threadIdx.x >> 5;
    const int hl   = lane >> 4;
    const int ln16 = lane & 15;
    const int m0   = (blockIdx.x * 4 + wave) << 4;   // this wave's 16-row tile
    const int nt0  = blockIdx.y * 4;                 // first of 4 n-tiles
    const int nK   = Kd >> 5;

    v8f acc0 = {}, acc1 = {}, acc2 = {}, acc3 = {};
    const _Float16* xrow = X + (size_t)(m0 + ln16) * Kd;
    const size_t tileStride = (size_t)nK * 512;      // halves between n-tiles in Wp

    for (int c = 0; c < nK; ++c) {
        // A fragment (ISA 16-bit A 16x32 layout): two contiguous b128 loads
        union { v16h v; v8h h2[2]; } A;
        A.h2[0] = *(const v8h*)(xrow + c * 32 + 8 * hl);
        A.h2[1] = *(const v8h*)(xrow + c * 32 + 16 + 8 * hl);
        if (c + 1 < nK)
            __builtin_prefetch(xrow + (c + 1) * 32, 0, 3);   // global_prefetch_b8

        // B fragments: pre-packed, two b128 loads per tile (coalesced 1KB/wave)
        const _Float16* bp = Wp + ((size_t)nt0 * nK + c) * 512 + (size_t)lane * 16;
        union { v16h v; v8h h2[2]; } B0, B1, B2, B3;
        B0.h2[0] = *(const v8h*)(bp);
        B0.h2[1] = *(const v8h*)(bp + 8);
        B1.h2[0] = *(const v8h*)(bp + tileStride);
        B1.h2[1] = *(const v8h*)(bp + tileStride + 8);
        B2.h2[0] = *(const v8h*)(bp + 2 * tileStride);
        B2.h2[1] = *(const v8h*)(bp + 2 * tileStride + 8);
        B3.h2[0] = *(const v8h*)(bp + 3 * tileStride);
        B3.h2[1] = *(const v8h*)(bp + 3 * tileStride + 8);

        acc0 = __builtin_amdgcn_wmma_f32_16x16x32_f16(false, A.v, false, B0.v,
                                                      (short)0, acc0, false, false);
        acc1 = __builtin_amdgcn_wmma_f32_16x16x32_f16(false, A.v, false, B1.v,
                                                      (short)0, acc1, false, false);
        acc2 = __builtin_amdgcn_wmma_f32_16x16x32_f16(false, A.v, false, B2.v,
                                                      (short)0, acc2, false, false);
        acc3 = __builtin_amdgcn_wmma_f32_16x16x32_f16(false, A.v, false, B3.v,
                                                      (short)0, acc3, false, false);
    }

    const float bv = bptr[0];
#pragma unroll
    for (int j = 0; j < 4; ++j) {
        const v8f& a = (j == 0) ? acc0 : (j == 1) ? acc1 : (j == 2) ? acc2 : acc3;
        int n = (nt0 + j) * 16 + ln16;
#pragma unroll
        for (int r = 0; r < 8; ++r) {
            int m = m0 + r + 8 * hl;     // C/D layout: VGPR r -> M=r (+8 for hi lanes)
            float o = a[r] + bv;
            Y[(size_t)m * Nd + n] = (_Float16)(o > 0.0f ? o : 0.0f);
        }
    }
}

// ---------------- 6: final layer (128 -> 1) + sigmoid ----------------
// one wave per batch row: 128-element dot with W3, relu, add wide term.
__global__ void daw_final(const _Float16* __restrict__ Y2,
                          const _Float16* __restrict__ W3h,
                          const float* __restrict__ b3,
                          const float* __restrict__ bias0,
                          const float* __restrict__ first,
                          float* __restrict__ out, int bsz) {
    int gid  = blockIdx.x * blockDim.x + threadIdx.x;
    int wid  = gid >> 5;            // one wave32 per row
    int lane = gid & 31;
    if (wid >= bsz) return;

    float s = 0.0f;
#pragma unroll
    for (int j = 0; j < N2; j += 32)
        s += (float)Y2[(size_t)wid * N2 + j + lane] * (float)W3h[j + lane];

    for (int off = 16; off > 0; off >>= 1)
        s += __shfl_down(s, off, 32);

    if (lane == 0) {
        float hi = s + b3[0];
        hi = hi > 0.0f ? hi : 0.0f;
        float z = first[wid] + bias0[0] + hi;
        out[wid] = 1.0f / (1.0f + __expf(-z));
    }
}

// ---------------- launch ----------------
extern "C" void kernel_launch(void* const* d_in, const int* in_sizes, int n_in,
                              void* d_out, int out_size, void* d_ws, size_t ws_size,
                              hipStream_t stream) {
    const int*   index     = (const int*)  d_in[1];
    const int*   feats     = (const int*)  d_in[2];
    const float* values    = (const float*)d_in[3];
    const int*   fields    = (const int*)  d_in[4];
    const float* bias      = (const float*)d_in[5];
    const float* weights   = (const float*)d_in[6];
    const float* embedding = (const float*)d_in[7];
    const float* W1        = (const float*)d_in[8];
    const float* b1        = (const float*)d_in[9];
    const float* W2        = (const float*)d_in[10];
    const float* b2        = (const float*)d_in[11];
    const float* W3        = (const float*)d_in[12];
    const float* b3        = (const float*)d_in[13];
    const int nnz = in_sizes[1];

    // workspace layout (256B aligned offsets); accumulators first and contiguous
    char* ws = (char*)d_ws;
    float* first = (float*)(ws + 0);                         // B floats
    float* cnt   = (float*)(ws + 65536);                     // B*NF floats
    float* h     = (float*)(ws + 3342336);                   // B*800 floats
    _Float16* X   = (_Float16*)(ws + 55771136);              // B*800 halves
    _Float16* Y1  = (_Float16*)(ws + 81985536);              // B*256 halves
    _Float16* Y2  = (_Float16*)(ws + 90374144);              // B*128 halves
    _Float16* W1p = (_Float16*)(ws + 94568448);              // 800*256 halves (packed)
    _Float16* W2p = (_Float16*)(ws + 94978048);              // 256*128 halves (packed)
    _Float16* W3h = (_Float16*)(ws + 95043584);              // 128 halves

    // 1: zero first|cnt|h (contiguous)
    const long nz_floats = (long)BSZ + (long)BSZ * NFLD + (long)BSZ * DIN;
    daw_zero<<<2048, 256, 0, stream>>>((float*)ws, nz_floats);

    // 3: weight packing / conversion
    daw_pack_b<<<(DIN * N1 + 255) / 256, 256, 0, stream>>>(W1, W1p, DIN, N1);
    daw_pack_b<<<(N1 * N2 + 255) / 256, 256, 0, stream>>>(W2, W2p, N1, N2);
    daw_cvt_f16<<<1, 256, 0, stream>>>(W3, W3h, N2);

    // 2: scatter
    long sc_threads = (long)nnz * EK;
    int sc_blocks = (int)((sc_threads + 255) / 256);
    daw_scatter<<<sc_blocks, 256, 0, stream>>>(index, feats, values, fields,
                                               weights, embedding,
                                               first, cnt, h, nnz);

    // 4: normalize -> f16 X
    long xt = (long)BSZ * DIN;
    daw_normalize<<<(int)((xt + 255) / 256), 256, 0, stream>>>(h, cnt, X, xt);

    // 5: MLP layers 1 and 2 via WMMA (4 waves/block, 16x64 strip per wave)
    {
        dim3 g1(BSZ / 64, N1 / 64);     // (256, 4)
        daw_gemm_relu_wmma<<<g1, 128, 0, stream>>>(X, W1p, b1, Y1, BSZ, DIN, N1);
        dim3 g2(BSZ / 64, N2 / 64);     // (256, 2)
        daw_gemm_relu_wmma<<<g2, 128, 0, stream>>>(Y1, W2p, b2, Y2, BSZ, N1, N2);
    }

    // 6: final dot + wide + sigmoid
    daw_final<<<(BSZ * 32 + 255) / 256, 256, 0, stream>>>(Y2, W3h, b3, bias,
                                                          first, (float*)d_out, BSZ);
}
